// context_branch_55825984914055
// MI455X (gfx1250) — matrix operand order repaired and checked by reading, not compiled, not run on hardware
//
#include <hip/hip_runtime.h>
#include <hip/hip_bf16.h>

// ---------------------------------------------------------------------------
// Deformable-DETR encoder+decoder layer for MI455X (gfx1250, wave32, WMMA).
// All dense GEMMs use v_wmma_f32_16x16x32_bf16 (fp32 activations/weights are
// converted to bf16 with native v_cvt during LDS staging; f32 accumulate).
// Input flattening assumption (setup_inputs dict order, params depth-first):
//  0..3 feat0..feat3, 4 context, 5 query_embed, 6 level_embed,
//  7..22 enc{so_w,so_b,aw_w,aw_b,vp_w,vp_b,op_w,op_b,ln1_w,ln1_b,fc1_w,fc1_b,
//            fc2_w,fc2_b,ln2_w,ln2_b},
// 23..44 dec{so_w,so_b,aw_w,aw_b,vp_w,vp_b,op_w,op_b,in_w,in_b,out_w,out_b,
//            ln1_w,ln1_b,ln2_w,ln2_b,ln3_w,ln3_b,fc1_w,fc1_b,fc2_w,fc2_b},
// 45 q_H, 46 q_W.
// ---------------------------------------------------------------------------

typedef __attribute__((ext_vector_type(16))) __bf16 bf16x16;
typedef __attribute__((ext_vector_type(8)))  float  f32x8;

#define DMODEL 256
#define NHEADS 8
#define DHEAD  32
#define SVAL   21760       // 128^2 + 64^2 + 32^2 + 16^2
#define LQTOT  4096
#define BSZ    2
#define TWO_PI 6.283185307179586f

__constant__ int c_lvl_off[4] = {0, 16384, 20480, 21504};

// Native f32 -> bf16 conversion (hardware RNE, lowers to v_cvt[_pk]_bf16_f32)
__device__ __forceinline__ unsigned pack2(float a, float b) {
  __bf16 x = (__bf16)a, y = (__bf16)b;
  return (unsigned)__builtin_bit_cast(unsigned short, x) |
         ((unsigned)__builtin_bit_cast(unsigned short, y) << 16);
}
union FragU { bf16x16 v; uint4 q[2]; unsigned short s[16]; };

// ---------------------------------------------------------------------------
// Tiled WMMA GEMM:  C[M,N] = act[M,K] @ W[N,K]^T + bias   (optional ReLU)
// Block tile 128x64, 8 waves (4x2), each wave 32x32 = 2x2 wmma frags, BK=32.
// M % 128 == 0, N % 64 == 0, K % 32 == 0 (holds for every GEMM in this net).
// ---------------------------------------------------------------------------
#define BM 128
#define BN 64
#define BK 32
#define LDSW 40   // padded LDS row stride (bf16 elems)

__global__ void __launch_bounds__(256)
wmma_gemm_bias(const float* __restrict__ A, const float* __restrict__ W,
               const float* __restrict__ bias, float* __restrict__ C,
               int M, int N, int K, int relu)
{
  __shared__ __attribute__((aligned(16))) __bf16 As[BM * LDSW];
  __shared__ __attribute__((aligned(16))) __bf16 Ws[BN * LDSW];
  const int tid  = threadIdx.x;
  const int m0   = blockIdx.y * BM;
  const int n0   = blockIdx.x * BN;
  const int wave = tid >> 5, lane = tid & 31;
  const int wm = wave & 3, wn = wave >> 2;
  const int lrow = lane & 15, lhalf = lane >> 4;

  f32x8 acc[2][2] = {};

  for (int k0 = 0; k0 < K; k0 += BK) {
    // stage A tile (128x32 fp32 -> bf16), 1024 float4 over 256 threads
    #pragma unroll
    for (int i = 0; i < 4; ++i) {
      int idx = tid + i * 256;
      int r = idx >> 3, c4 = idx & 7;
      float4 v = *(const float4*)(A + (size_t)(m0 + r) * K + k0 + c4 * 4);
      uint2 p; p.x = pack2(v.x, v.y); p.y = pack2(v.z, v.w);
      *(uint2*)(As + r * LDSW + c4 * 4) = p;
    }
    // stage W tile (64x32), 512 float4
    #pragma unroll
    for (int i = 0; i < 2; ++i) {
      int idx = tid + i * 256;
      int r = idx >> 3, c4 = idx & 7;
      float4 v = *(const float4*)(W + (size_t)(n0 + r) * K + k0 + c4 * 4);
      uint2 p; p.x = pack2(v.x, v.y); p.y = pack2(v.z, v.w);
      *(uint2*)(Ws + r * LDSW + c4 * 4) = p;
    }
    if (k0 + BK < K) {   // gfx1250 global_prefetch_b8 hint for next K tile
      int r = tid >> 3, c4 = tid & 7;
      __builtin_prefetch(A + (size_t)(m0 + r) * K + (k0 + BK) + c4 * 4, 0, 1);
    }
    __syncthreads();

    FragU afr[2], bfr[2];
    #pragma unroll
    for (int i = 0; i < 2; ++i) {       // A frag: halves split at K=8
      int r = wm * 32 + i * 16 + lrow;
      afr[i].q[0] = *(const uint4*)(As + r * LDSW + lhalf * 8);
      afr[i].q[1] = *(const uint4*)(As + r * LDSW + 16 + lhalf * 8);
    }
    #pragma unroll
    for (int j = 0; j < 2; ++j) {       // B frag: halves split at K=8
      int r = wn * 32 + j * 16 + lrow;
      bfr[j].q[0] = *(const uint4*)(Ws + r * LDSW + lhalf * 8);
      bfr[j].q[1] = *(const uint4*)(Ws + r * LDSW + 16 + lhalf * 8);
    }
    #pragma unroll
    for (int i = 0; i < 2; ++i)
      #pragma unroll
      for (int j = 0; j < 2; ++j)
        acc[i][j] = __builtin_amdgcn_wmma_f32_16x16x32_bf16(
            false, afr[i].v, false, bfr[j].v, (short)0, acc[i][j], false, false);
    __syncthreads();
  }

  #pragma unroll
  for (int i = 0; i < 2; ++i)
    #pragma unroll
    for (int j = 0; j < 2; ++j) {
      int col = n0 + wn * 32 + j * 16 + lrow;
      float bv = bias ? bias[col] : 0.f;
      #pragma unroll
      for (int r = 0; r < 8; ++r) {
        int row = m0 + wm * 32 + i * 16 + r + 8 * lhalf;
        float v = acc[i][j][r] + bv;
        if (relu) v = fmaxf(v, 0.f);
        C[(size_t)row * N + col] = v;
      }
    }
}

// ---------------------------------------------------------------------------
// Build src (feat transpose) + qsrc = src + sine pos embed + level embed.
// grid = (SVAL, BSZ), block = 256 (one channel per thread).
// ---------------------------------------------------------------------------
__global__ void __launch_bounds__(256)
build_src_pos(const float* __restrict__ f0, const float* __restrict__ f1,
              const float* __restrict__ f2, const float* __restrict__ f3,
              const float* __restrict__ lvl_emb,
              float* __restrict__ src, float* __restrict__ qsrc)
{
  int s = blockIdx.x, b = blockIdx.y, c = threadIdx.x;
  int l = (s >= 16384) + (s >= 20480) + (s >= 21504);
  int Wl = 128 >> l;                       // levels are square
  int sl = s - c_lvl_off[l];
  int i = sl / Wl, j = sl % Wl;
  const float* f = (l == 0) ? f0 : (l == 1) ? f1 : (l == 2) ? f2 : f3;
  float val = f[(((size_t)b * DMODEL + c) * Wl + i) * Wl + j];

  int cc = c & 127;
  float v = (c < 128) ? ((float)(i + 1) / ((float)Wl + 1e-6f) * TWO_PI)
                      : ((float)(j + 1) / ((float)Wl + 1e-6f) * TWO_PI);
  float t = __powf(10000.f, (float)(cc >> 1) * (2.0f / 128.0f));
  float p = v / t;
  float pe = (cc & 1) ? __cosf(p) : __sinf(p);
  pe += lvl_emb[l * DMODEL + c];

  size_t idx = ((size_t)b * SVAL + s) * DMODEL + c;
  src[idx]  = val;
  qsrc[idx] = val + pe;
}

// ---------------------------------------------------------------------------
// Multi-scale deformable attention core: one wave per (b,q,head); lane = dh.
// refmode 0: encoder (ref from concatenated level grids), 1: decoder (qH x qW).
// grid = BSZ*Q blocks of 256 (8 waves = 8 heads).
// ---------------------------------------------------------------------------
__device__ __forceinline__ float bil_tap(const float* __restrict__ vb,
                                         int xi, int yi, int Wl, int Hl) {
  if (xi < 0 || yi < 0 || xi >= Wl || yi >= Hl) return 0.f;
  return vb[(size_t)(yi * Wl + xi) * DMODEL];
}

__global__ void __launch_bounds__(256)
msdeform_core(const float* __restrict__ value, const float* __restrict__ soff,
              const float* __restrict__ aw, float* __restrict__ out,
              int Q, int refmode,
              const int* __restrict__ qWp, const int* __restrict__ qHp)
{
  int h = threadIdx.x >> 5, lane = threadIdx.x & 31;
  int bq = blockIdx.x;
  int q = bq % Q, b = bq / Q;

  float rx, ry;
  if (refmode == 0) {
    int l = (q >= 16384) + (q >= 20480) + (q >= 21504);
    int Wl = 128 >> l;
    int sl = q - c_lvl_off[l];
    int i = sl / Wl, j = sl % Wl;
    rx = ((float)j + 0.5f) / (float)Wl;
    ry = ((float)i + 0.5f) / (float)Wl;
  } else {
    int qW = *qWp, qH = *qHp;
    int i = q / qW, j = q % qW;
    rx = ((float)j + 0.5f) / (float)qW;
    ry = ((float)i + 0.5f) / (float)qH;
  }

  size_t row = (size_t)b * Q + q;
  const float* ap = aw + row * 128 + h * 16;       // wave-uniform -> scalar loads
  float lg[16]; float mx = -1e30f;
  #pragma unroll
  for (int k = 0; k < 16; ++k) { lg[k] = ap[k]; mx = fmaxf(mx, lg[k]); }
  float se = 0.f;
  #pragma unroll
  for (int k = 0; k < 16; ++k) { lg[k] = __expf(lg[k] - mx); se += lg[k]; }
  float inv = 1.f / se;

  const float* op = soff + row * DMODEL + h * DHEAD;
  float acc = 0.f;
  #pragma unroll
  for (int l = 0; l < 4; ++l) {
    int Wl = 128 >> l;
    const float* vbase = value + ((size_t)b * SVAL + c_lvl_off[l]) * DMODEL + h * DHEAD + lane;
    #pragma unroll
    for (int p = 0; p < 4; ++p) {
      float ox = op[l * 8 + p * 2], oy = op[l * 8 + p * 2 + 1];
      float lx = rx + ox / (float)Wl;
      float ly = ry + oy / (float)Wl;
      float xf = lx * (float)Wl - 0.5f, yf = ly * (float)Wl - 0.5f;
      float x0f = floorf(xf), y0f = floorf(yf);
      int x0 = (int)x0f, y0 = (int)y0f;
      float wx = xf - x0f, wy = yf - y0f;
      float a = lg[l * 4 + p] * inv;
      float t00 = bil_tap(vbase, x0,     y0,     Wl, Wl);
      float t10 = bil_tap(vbase, x0 + 1, y0,     Wl, Wl);
      float t01 = bil_tap(vbase, x0,     y0 + 1, Wl, Wl);
      float t11 = bil_tap(vbase, x0 + 1, y0 + 1, Wl, Wl);
      acc += a * (t00 * (1.f - wx) * (1.f - wy) + t10 * wx * (1.f - wy)
                + t01 * (1.f - wx) * wy         + t11 * wx * wy);
    }
  }
  out[row * DMODEL + h * DHEAD + lane] = acc;
}

// ---------------------------------------------------------------------------
// residual add + LayerNorm over 256 channels; one row per block.
// transpose_out: write [Lq, bs, d] layout (rows come in as b*LQ+q).
// ---------------------------------------------------------------------------
__global__ void __launch_bounds__(256)
add_ln(const float* __restrict__ x, const float* __restrict__ y,
       const float* __restrict__ w, const float* __restrict__ b,
       float* __restrict__ out, int transpose_out)
{
  __shared__ float sm[8], sq[8];
  int row = blockIdx.x, c = threadIdx.x;
  size_t idx = (size_t)row * DMODEL + c;
  float v = x[idx] + y[idx];
  float s = v, s2 = v * v;
  #pragma unroll
  for (int o = 1; o < 32; o <<= 1) {
    s  += __shfl_xor(s,  o, 32);
    s2 += __shfl_xor(s2, o, 32);
  }
  if ((c & 31) == 0) { sm[c >> 5] = s; sq[c >> 5] = s2; }
  __syncthreads();
  float ts = 0.f, tq = 0.f;
  #pragma unroll
  for (int i = 0; i < 8; ++i) { ts += sm[i]; tq += sq[i]; }
  float mean = ts * (1.f / 256.f);
  float var  = tq * (1.f / 256.f) - mean * mean;
  float rs = rsqrtf(var + 1e-5f);
  float o = (v - mean) * rs * w[c] + b[c];
  if (transpose_out) {
    int bb = row >> 12, q = row & 4095;
    out[((size_t)q * BSZ + bb) * DMODEL + c] = o;
  } else {
    out[idx] = o;
  }
}

// tgtb[b*LQ+q][c] = context[q][b][c] + query_embed[q][c]
__global__ void __launch_bounds__(256)
build_tgt(const float* __restrict__ ctx, const float* __restrict__ qe,
          float* __restrict__ tgtb)
{
  int c = threadIdx.x, r = blockIdx.x;       // r = b*4096 + q
  int b = r >> 12, q = r & 4095;
  tgtb[(size_t)r * DMODEL + c] =
      ctx[((size_t)q * BSZ + b) * DMODEL + c] + qe[(size_t)q * DMODEL + c];
}

// ---------------------------------------------------------------------------
// Decoder self-attention. The reference (faithful to the source transpose bug)
// attends over L = bs = 2 with batch = Lq queries: per (q,h) a 2x2 softmax.
// One wave per (q,h); lane = dh. grid = LQ*H/8 blocks of 256.
// ---------------------------------------------------------------------------
__global__ void __launch_bounds__(256)
mha_len2(const float* __restrict__ qkvp, float* __restrict__ out)
{
  int wave = threadIdx.x >> 5, lane = threadIdx.x & 31;
  int wid = blockIdx.x * 8 + wave;
  int h = wid & 7, q = wid >> 3;
  const float scale = 0.17677669529663687f;   // 32^-0.5
  size_t r0 = (size_t)q * 768, r1 = (size_t)(LQTOT + q) * 768;
  int c = h * DHEAD + lane;
  float q0 = qkvp[r0 + c] * scale, q1 = qkvp[r1 + c] * scale;
  float k0 = qkvp[r0 + 256 + c],   k1 = qkvp[r1 + 256 + c];
  float v0 = qkvp[r0 + 512 + c],   v1 = qkvp[r1 + 512 + c];
  float d00 = q0 * k0, d01 = q0 * k1, d10 = q1 * k0, d11 = q1 * k1;
  #pragma unroll
  for (int o = 1; o < 32; o <<= 1) {
    d00 += __shfl_xor(d00, o, 32); d01 += __shfl_xor(d01, o, 32);
    d10 += __shfl_xor(d10, o, 32); d11 += __shfl_xor(d11, o, 32);
  }
  float m0 = fmaxf(d00, d01), m1 = fmaxf(d10, d11);
  float e00 = __expf(d00 - m0), e01 = __expf(d01 - m0);
  float e10 = __expf(d10 - m1), e11 = __expf(d11 - m1);
  float i0 = 1.f / (e00 + e01), i1 = 1.f / (e10 + e11);
  out[(size_t)q * DMODEL + c]           = (e00 * v0 + e01 * v1) * i0;
  out[((size_t)LQTOT + q) * DMODEL + c] = (e10 * v0 + e11 * v1) * i1;
}

// ---------------------------------------------------------------------------
static inline void gemm(const float* A, const float* W, const float* b, float* C,
                        int M, int N, int K, int relu, hipStream_t s) {
  dim3 g(N / BN, M / BM);
  wmma_gemm_bias<<<g, 256, 0, s>>>(A, W, b, C, M, N, K, relu);
}

extern "C" void kernel_launch(void* const* d_in, const int* in_sizes, int n_in,
                              void* d_out, int out_size, void* d_ws, size_t ws_size,
                              hipStream_t stream)
{
  if (n_in < 47) return;
  const float* f0 = (const float*)d_in[0];
  const float* f1 = (const float*)d_in[1];
  const float* f2 = (const float*)d_in[2];
  const float* f3 = (const float*)d_in[3];
  const float* ctx = (const float*)d_in[4];
  const float* qemb = (const float*)d_in[5];
  const float* lvl_emb = (const float*)d_in[6];
  const float *e_so_w=(const float*)d_in[7],  *e_so_b=(const float*)d_in[8];
  const float *e_aw_w=(const float*)d_in[9],  *e_aw_b=(const float*)d_in[10];
  const float *e_vp_w=(const float*)d_in[11], *e_vp_b=(const float*)d_in[12];
  const float *e_op_w=(const float*)d_in[13], *e_op_b=(const float*)d_in[14];
  const float *e_ln1w=(const float*)d_in[15], *e_ln1b=(const float*)d_in[16];
  const float *e_fc1w=(const float*)d_in[17], *e_fc1b=(const float*)d_in[18];
  const float *e_fc2w=(const float*)d_in[19], *e_fc2b=(const float*)d_in[20];
  const float *e_ln2w=(const float*)d_in[21], *e_ln2b=(const float*)d_in[22];
  const float *k_so_w=(const float*)d_in[23], *k_so_b=(const float*)d_in[24];
  const float *k_aw_w=(const float*)d_in[25], *k_aw_b=(const float*)d_in[26];
  const float *k_vp_w=(const float*)d_in[27], *k_vp_b=(const float*)d_in[28];
  const float *k_op_w=(const float*)d_in[29], *k_op_b=(const float*)d_in[30];
  const float *k_in_w=(const float*)d_in[31], *k_in_b=(const float*)d_in[32];
  const float *k_out_w=(const float*)d_in[33],*k_out_b=(const float*)d_in[34];
  const float *k_ln1w=(const float*)d_in[35], *k_ln1b=(const float*)d_in[36];
  const float *k_ln2w=(const float*)d_in[37], *k_ln2b=(const float*)d_in[38];
  const float *k_ln3w=(const float*)d_in[39], *k_ln3b=(const float*)d_in[40];
  const float *k_fc1w=(const float*)d_in[41], *k_fc1b=(const float*)d_in[42];
  const float *k_fc2w=(const float*)d_in[43], *k_fc2b=(const float*)d_in[44];
  const int* qHp = (const int*)d_in[45];
  const int* qWp = (const int*)d_in[46];

  // workspace arena (floats), reuse-planned
  const size_t BIGN = (size_t)BSZ * SVAL * DMODEL;   // 11,141,120
  if (ws_size < (6 * BIGN + (size_t)BSZ * SVAL * 1024) * sizeof(float)) return;
  float* ws = (float*)d_ws;
  float* srcb  = ws + 0 * BIGN;   // B0: src -> later memory
  float* qsrc  = ws + 1 * BIGN;   // B1: src+pos -> later attnout
  float* value = ws + 2 * BIGN;   // B2: enc value -> later dec value
  float* soff  = ws + 3 * BIGN;   // B3: enc offsets -> later src2norm
  float* awraw = ws + 4 * BIGN;   // B4: enc aw logits -> later ffnout
  float* msout = ws + 5 * BIGN;   // B5: deform output
  float* hbuf  = ws + 6 * BIGN;   // 44.6M floats: enc FFN hidden, then decoder
  float* attnout = qsrc;
  float* src2n   = soff;
  float* h1      = hbuf;
  float* ffnout  = awraw;
  float* memry   = srcb;
  float* value2  = value;
  float* tgtb    = hbuf;
  float* qkvp    = hbuf + 2097152;
  float* mhaout  = hbuf + 8388608;
  float* mhaproj = hbuf + 10485760;
  float* tgt2    = hbuf + 12582912;
  float* soff2   = hbuf + 14680064;
  float* aw2     = hbuf + 16777216;
  float* msout2  = hbuf + 17825792;
  float* cross   = hbuf + 19922944;
  float* tgt3    = hbuf + 22020096;
  float* h2      = hbuf + 24117248;
  float* f2buf   = hbuf + 32505856;

  const int Me = BSZ * SVAL;   // 43520
  const int Md = BSZ * LQTOT;  // 8192

  // -------- encoder --------
  build_src_pos<<<dim3(SVAL, BSZ), 256, 0, stream>>>(f0, f1, f2, f3, lvl_emb, srcb, qsrc);
  gemm(srcb, e_vp_w, e_vp_b, value, Me, 256, 256, 0, stream);
  gemm(qsrc, e_so_w, e_so_b, soff,  Me, 256, 256, 0, stream);
  gemm(qsrc, e_aw_w, e_aw_b, awraw, Me, 128, 256, 0, stream);
  msdeform_core<<<Me, 256, 0, stream>>>(value, soff, awraw, msout, SVAL, 0, qWp, qHp);
  gemm(msout, e_op_w, e_op_b, attnout, Me, 256, 256, 0, stream);
  add_ln<<<Me, 256, 0, stream>>>(srcb, attnout, e_ln1w, e_ln1b, src2n, 0);
  gemm(src2n, e_fc1w, e_fc1b, h1,     Me, 1024, 256, 1, stream);
  gemm(h1,    e_fc2w, e_fc2b, ffnout, Me, 256, 1024, 0, stream);
  add_ln<<<Me, 256, 0, stream>>>(src2n, ffnout, e_ln2w, e_ln2b, memry, 0);

  // -------- decoder --------
  build_tgt<<<Md, 256, 0, stream>>>(ctx, qemb, tgtb);
  gemm(tgtb, k_in_w, k_in_b, qkvp, Md, 768, 256, 0, stream);
  mha_len2<<<LQTOT * NHEADS / 8, 256, 0, stream>>>(qkvp, mhaout);
  gemm(mhaout, k_out_w, k_out_b, mhaproj, Md, 256, 256, 0, stream);
  add_ln<<<Md, 256, 0, stream>>>(tgtb, mhaproj, k_ln2w, k_ln2b, tgt2, 0);
  gemm(memry, k_vp_w, k_vp_b, value2, Me, 256, 256, 0, stream);
  gemm(tgt2,  k_so_w, k_so_b, soff2,  Md, 256, 256, 0, stream);
  gemm(tgt2,  k_aw_w, k_aw_b, aw2,    Md, 128, 256, 0, stream);
  msdeform_core<<<Md, 256, 0, stream>>>(value2, soff2, aw2, msout2, LQTOT, 1, qWp, qHp);
  gemm(msout2, k_op_w, k_op_b, cross, Md, 256, 256, 0, stream);
  add_ln<<<Md, 256, 0, stream>>>(tgt2, cross, k_ln1w, k_ln1b, tgt3, 0);
  gemm(tgt3, k_fc1w, k_fc1b, h2,    Md, 1024, 256, 1, stream);
  gemm(h2,   k_fc2w, k_fc2b, f2buf, Md, 256, 1024, 0, stream);
  add_ln<<<Md, 256, 0, stream>>>(tgt3, f2buf, k_ln3w, k_ln3b, (float*)d_out, 1);
}
